// SaliencyRegionDetector_53919019434152
// MI455X (gfx1250) — matrix operand order repaired and checked by reading, not compiled, not run on hardware
//
#include <hip/hip_runtime.h>
#include <hip/hip_bf16.h>
#include <math.h>

// ---------------------------------------------------------------------------
// SaliencyRegionDetector for MI455X (gfx1250, wave32, WMMA + async-to-LDS)
//   stage0: gray + 5x5 box + sobel -> mag (+ global max via uint atomics)
//   stage1: 32^3 histogram with 128KB LDS-private histogram (CDNA5 320KB LDS)
//   stage2: band threshold -> f16 edge map
//   stage3: 15-tap horizontal Gaussian: async-LDS tile + 1 v_wmma / 16x16 tile
//   stage4: 15-tap vertical   Gaussian: async-LDS tile + 1 v_wmma / 16x16 tile
//   stage5: cs min/max derived from histogram (monotone in count)
//   stage6: fused normalize + sigmoid (with global_prefetch)
// ---------------------------------------------------------------------------

typedef __attribute__((ext_vector_type(16))) _Float16 v16h;
typedef __attribute__((ext_vector_type(8)))  float    v8f;

#define WIMG 3072
#define HIMG 3072
#define NPIX (3072 * 3072)
#define TILES_X (3072 / 16)

// Normalized 1-D Gaussian, size 15, sigma 2 (2D kernel = outer product)
__device__ __constant__ float GW15[15] = {
    0.000436408f, 0.002216256f, 0.008765477f, 0.027000580f, 0.064768620f,
    0.121003716f, 0.176059330f, 0.199501366f, 0.176059330f, 0.121003716f,
    0.064768620f, 0.027000580f, 0.008765477f, 0.002216256f, 0.000436408f};

// Async global->LDS copy of one aligned dword (2 f16), ASYNCcnt-tracked.
__device__ __forceinline__ void async_g2l_b32(const void* gaddr, void* lds_generic) {
    unsigned ldsoff = (unsigned)(uintptr_t)lds_generic;  // low 32b = LDS offset
    asm volatile("global_load_async_to_lds_b32 %0, %1, off"
                 :: "v"(ldsoff), "v"((unsigned long long)(uintptr_t)gaddr)
                 : "memory");
}
__device__ __forceinline__ void wait_asynccnt0() {
    asm volatile("s_wait_asynccnt 0x0" ::: "memory");
}

// scal[] layout: 0=magmax_bits 1=edmin_bits 2=edmax_bits 3=minPosCount 4=maxCount
__global__ void k_init(unsigned* __restrict__ hist, unsigned* __restrict__ scal) {
    int i = blockIdx.x * blockDim.x + threadIdx.x;
    if (i < 32768) hist[i] = 0u;
    if (i == 0) {
        scal[0] = 0u;
        scal[1] = 0x7F7FFFFFu;   // +FLT_MAX bits
        scal[2] = 0u;
        scal[3] = 0xFFFFFFFFu;
        scal[4] = 0u;
    }
}

// gray -> 5x5 box -> sobel -> mag, fused per 16x16 tile with LDS halos
__global__ void k_gray_box_sobel(const float* __restrict__ img,
                                 float* __restrict__ mag,
                                 unsigned* __restrict__ magmax_bits) {
    __shared__ float gsh[22 * 22];     // gray, halo 3
    __shared__ float bsh[18 * 18];     // box-blurred, halo 1
    __shared__ unsigned smax;
    const int tx = blockIdx.x % TILES_X;
    const int ty = blockIdx.x / TILES_X;
    const int x0 = tx * 16, y0 = ty * 16;
    const int t = threadIdx.x;  // 256 threads
    if (t == 0) smax = 0u;
    for (int m = t; m < 22 * 22; m += 256) {
        int r = m / 22, c = m % 22;
        int gy = y0 - 3 + r, gx = x0 - 3 + c;
        float v = 0.f;
        if (gx >= 0 && gx < WIMG && gy >= 0 && gy < HIMG) {
            size_t p = (size_t)gy * WIMG + gx;
            v = 0.299f * img[p] + 0.587f * img[p + (size_t)NPIX] +
                0.114f * img[p + 2 * (size_t)NPIX];
        }
        gsh[m] = v;
    }
    __syncthreads();
    for (int m = t; m < 18 * 18; m += 256) {
        int r = m / 18, c = m % 18;
        float s = 0.f;
#pragma unroll
        for (int dr = 0; dr < 5; ++dr)
#pragma unroll
            for (int dc = 0; dc < 5; ++dc) s += gsh[(r + dr) * 22 + (c + dc)];
        bsh[m] = s * (1.0f / 25.0f);
    }
    __syncthreads();
    {
        int r = t >> 4, c = t & 15;  // output pixel; blurred tile idx (r+1,c+1)
        float b00 = bsh[(r + 0) * 18 + c + 0], b01 = bsh[(r + 0) * 18 + c + 1],
              b02 = bsh[(r + 0) * 18 + c + 2];
        float b10 = bsh[(r + 1) * 18 + c + 0], b12 = bsh[(r + 1) * 18 + c + 2];
        float b20 = bsh[(r + 2) * 18 + c + 0], b21 = bsh[(r + 2) * 18 + c + 1],
              b22 = bsh[(r + 2) * 18 + c + 2];
        float gx = -b00 + b02 - 2.f * b10 + 2.f * b12 - b20 + b22;
        float gy = -b00 - 2.f * b01 - b02 + b20 + 2.f * b21 + b22;
        float mv = sqrtf(gx * gx + gy * gy);
        mag[(size_t)(y0 + r) * WIMG + x0 + c] = mv;
        atomicMax(&smax, __float_as_uint(mv));  // mv >= 0: bit order == value order
    }
    __syncthreads();
    if (t == 0) atomicMax(magmax_bits, smax);
}

// full 32^3 histogram privatized in 128KB LDS (needs CDNA5's 320KB WGP LDS)
__global__ void k_hist(const float* __restrict__ img, unsigned* __restrict__ hist) {
    extern __shared__ unsigned lh[];  // 32768 bins
    const int t = threadIdx.x;        // 1024 threads
    for (int b = t; b < 32768; b += 1024) lh[b] = 0u;
    __syncthreads();
    const size_t stride = (size_t)gridDim.x * blockDim.x;
    for (size_t p = (size_t)blockIdx.x * blockDim.x + t; p < NPIX; p += stride) {
        int r  = (int)fminf(fmaxf(rintf(img[p] * 31.f), 0.f), 31.f);
        int g  = (int)fminf(fmaxf(rintf(img[p + (size_t)NPIX] * 31.f), 0.f), 31.f);
        int bl = (int)fminf(fmaxf(rintf(img[p + 2 * (size_t)NPIX] * 31.f), 0.f), 31.f);
        atomicAdd(&lh[r * 1024 + g * 32 + bl], 1u);
    }
    __syncthreads();
    for (int b = t; b < 32768; b += 1024) {
        unsigned c = lh[b];
        if (c) atomicAdd(&hist[b], c);
    }
}

__global__ void k_threshold(const float* __restrict__ mag,
                            const unsigned* __restrict__ magmax_bits,
                            _Float16* __restrict__ edge) {
    const float hi = __uint_as_float(*magmax_bits) * 0.2f;
    const float lo = hi * 0.3f;
    const size_t stride = (size_t)gridDim.x * blockDim.x;
    for (size_t p = (size_t)blockIdx.x * blockDim.x + threadIdx.x; p < NPIX; p += stride) {
        float m = mag[p];
        edge[p] = (_Float16)((m > lo && m < hi) ? 1.0f : 0.0f);
    }
}

// ISA 7.12.2 16-bit A-matrix interleave: element e of the v16h -> K index
__device__ __forceinline__ int a_kidx(int e, int half) {
    int v = e >> 1;
    return ((v & 4) ? 16 : 0) + half * 8 + (v & 3) * 2 + (e & 1);
}

// horizontal 15-tap Gaussian: Out(16x16) = A(16x32 data) x W(32x16 banded)
// Tile cols x0-8 .. x0+23 (dword-aligned pairs; tap t = K - n - 1)
__global__ void k_gauss_h(const _Float16* __restrict__ src, _Float16* __restrict__ dst) {
    __shared__ _Float16 sh[4][16 * 32];
    const int lane = threadIdx.x & 31;
    const int wv = threadIdx.x >> 5;
    const int tile = blockIdx.x * 4 + wv;
    const int tx = tile % TILES_X, ty = tile / TILES_X;
    const int x0 = tx * 16, y0 = ty * 16;
    _Float16* s = sh[wv];
#pragma unroll
    for (int i = 0; i < 8; ++i) {  // 256 dword pairs: 16 rows x 16 pairs
        int q = i * 32 + lane;
        int r = q >> 4, c2 = (q & 15) * 2;
        int gx = x0 - 8 + c2;      // even => dword aligned; pairs never straddle edge
        if (gx >= 0 && gx < WIMG) {
            async_g2l_b32(src + (size_t)(y0 + r) * WIMG + gx, s + (r * 32 + c2));
        } else {
            *(unsigned*)(s + (r * 32 + c2)) = 0u;  // disjoint from async targets
        }
    }
    wait_asynccnt0();
    __syncthreads();
    const int half = lane >> 4, n = lane & 15;
    v16h a, b;
#pragma unroll
    for (int e = 0; e < 16; ++e) {
        int kk = a_kidx(e, half);
        a[e] = s[n * 32 + kk];          // A[M=n][K=kk]
        int K = kk;                     // B[K][N=n] = g1[K-n-1] (banded)
        int d = K - n - 1;
        b[e] = (d >= 0 && d < 15) ? (_Float16)GW15[d] : (_Float16)0.f;
    }
    v8f acc = {};
    acc = __builtin_amdgcn_wmma_f32_16x16x32_f16(false, a, false, b, (short)0, acc,
                                                 false, false);
#pragma unroll
    for (int v = 0; v < 8; ++v) {
        int M = v + 8 * half;
        dst[(size_t)(y0 + M) * WIMG + x0 + n] = (_Float16)acc[v];
    }
}

// vertical 15-tap Gaussian: Out(16x16) = W(16x32 banded) x B(32x16 data)
// Tile rows y0-8 .. y0+23 (tap t = K - n - 1)
__global__ void k_gauss_v(const _Float16* __restrict__ src, float* __restrict__ ed,
                          unsigned* __restrict__ edmin_bits,
                          unsigned* __restrict__ edmax_bits) {
    __shared__ _Float16 sh[4][32 * 16];
    const int lane = threadIdx.x & 31;
    const int wv = threadIdx.x >> 5;
    const int tile = blockIdx.x * 4 + wv;
    const int tx = tile % TILES_X, ty = tile / TILES_X;
    const int x0 = tx * 16, y0 = ty * 16;
    _Float16* s = sh[wv];
#pragma unroll
    for (int i = 0; i < 8; ++i) {  // 256 dword pairs: 32 rows x 8 pairs
        int q = i * 32 + lane;
        int r = q >> 3, c2 = (q & 7) * 2;
        int gy = y0 - 8 + r;
        if (gy >= 0 && gy < HIMG) {
            async_g2l_b32(src + (size_t)gy * WIMG + x0 + c2, s + (r * 16 + c2));
        } else {
            *(unsigned*)(s + (r * 16 + c2)) = 0u;
        }
    }
    wait_asynccnt0();
    __syncthreads();
    const int half = lane >> 4, n = lane & 15;
    v16h a, b;
#pragma unroll
    for (int e = 0; e < 16; ++e) {
        int kk = a_kidx(e, half);
        int d = kk - n - 1;             // A[M=n][K=kk] = g1[kk-n-1] (banded)
        a[e] = (d >= 0 && d < 15) ? (_Float16)GW15[d] : (_Float16)0.f;
        int K = kk;                     // B[K][N=n] = tile row K, col n
        b[e] = s[K * 16 + n];
    }
    v8f acc = {};
    acc = __builtin_amdgcn_wmma_f32_16x16x32_f16(false, a, false, b, (short)0, acc,
                                                 false, false);
    float lmin = 3.4e38f, lmax = 0.f;
#pragma unroll
    for (int v = 0; v < 8; ++v) {
        int M = v + 8 * half;
        float val = acc[v];
        ed[(size_t)(y0 + M) * WIMG + x0 + n] = val;
        lmin = fminf(lmin, val);
        lmax = fmaxf(lmax, val);
    }
    atomicMin(edmin_bits, __float_as_uint(lmin));  // ed >= 0
    atomicMax(edmax_bits, __float_as_uint(lmax));
}

// cs is monotone-decreasing in hist count -> min/max from histogram alone
__global__ void k_csstats(const unsigned* __restrict__ hist,
                          unsigned* __restrict__ maxcnt,
                          unsigned* __restrict__ minpos) {
    unsigned lmax = 0u, lmin = 0xFFFFFFFFu;
    const int stride = gridDim.x * blockDim.x;
    for (int b = blockIdx.x * blockDim.x + threadIdx.x; b < 32768; b += stride) {
        unsigned c = hist[b];
        if (c) {
            lmax = (c > lmax) ? c : lmax;
            lmin = (c < lmin) ? c : lmin;
        }
    }
    atomicMax(maxcnt, lmax);
    atomicMin(minpos, lmin);
}

__global__ void k_final(const float* __restrict__ img, const float* __restrict__ ed,
                        const unsigned* __restrict__ hist,
                        const unsigned* __restrict__ scal,
                        const float* __restrict__ alpha, const float* __restrict__ beta,
                        float* __restrict__ out) {
    const float edmin = __uint_as_float(scal[1]);
    const float edmax = __uint_as_float(scal[2]);
    const float invN = 1.0f / (float)NPIX;
    const float csmin = -logf((float)scal[4] * invN + 1e-9f) * 1.5f;  // max count
    const float csmax = -logf((float)scal[3] * invN + 1e-9f) * 1.5f;  // min pos count
    const float al = *alpha, be = *beta;
    const float edscale = 1.0f / (edmax - edmin + 1e-9f);
    const float csscale = 1.0f / (csmax - csmin + 1e-9f);
    const size_t stride = (size_t)gridDim.x * blockDim.x;
    for (size_t p = (size_t)blockIdx.x * blockDim.x + threadIdx.x; p < NPIX; p += stride) {
        size_t pn = p + stride;
        if (pn < NPIX) {  // stream-ahead prefetch -> global_prefetch_b8
            __builtin_prefetch(&img[pn], 0, 1);
            __builtin_prefetch(&ed[pn], 0, 1);
        }
        int r  = (int)fminf(fmaxf(rintf(img[p] * 31.f), 0.f), 31.f);
        int g  = (int)fminf(fmaxf(rintf(img[p + (size_t)NPIX] * 31.f), 0.f), 31.f);
        int bl = (int)fminf(fmaxf(rintf(img[p + 2 * (size_t)NPIX] * 31.f), 0.f), 31.f);
        float cnt = (float)hist[r * 1024 + g * 32 + bl];
        float cs = -logf(cnt * invN + 1e-9f) * 1.5f;
        float en = (ed[p] - edmin) * edscale;
        float cn = (cs - csmin) * csscale;
        float z = al * en + be * cn;
        out[p] = 1.0f / (1.0f + __expf(-z));
    }
}

extern "C" void kernel_launch(void* const* d_in, const int* in_sizes, int n_in,
                              void* d_out, int out_size, void* d_ws, size_t ws_size,
                              hipStream_t stream) {
    const float* img   = (const float*)d_in[0];
    const float* alpha = (const float*)d_in[1];
    const float* beta  = (const float*)d_in[2];
    float* out = (float*)d_out;

    // workspace layout (~75.6 MB):
    //   [0,4N)      bufA f32 : mag, reused as ed after thresholding
    //   [4N,6N)     bufB f16 : edge map
    //   [6N,8N)     bufC f16 : horizontal-blur temp
    //   [8N,+128K)  hist u32[32768]
    //   then        scal u32[8]
    char* ws = (char*)d_ws;
    const size_t N = (size_t)NPIX;
    float*    bufA = (float*)ws;
    _Float16* bufB = (_Float16*)(ws + 4 * N);
    _Float16* bufC = (_Float16*)(ws + 6 * N);
    unsigned* hist = (unsigned*)(ws + 8 * N);
    unsigned* scal = (unsigned*)(ws + 8 * N + 32768 * sizeof(unsigned));

    k_init<<<128, 256, 0, stream>>>(hist, scal);
    k_gray_box_sobel<<<TILES_X * TILES_X, 256, 0, stream>>>(img, bufA, &scal[0]);
    k_hist<<<512, 1024, 32768 * sizeof(unsigned), stream>>>(img, hist);
    k_threshold<<<4608, 256, 0, stream>>>(bufA, &scal[0], bufB);
    k_gauss_h<<<TILES_X * TILES_X / 4, 128, 0, stream>>>(bufB, bufC);
    k_gauss_v<<<TILES_X * TILES_X / 4, 128, 0, stream>>>(bufC, bufA, &scal[1], &scal[2]);
    k_csstats<<<32, 256, 0, stream>>>(hist, &scal[4], &scal[3]);
    k_final<<<4608, 256, 0, stream>>>(img, bufA, hist, scal, alpha, beta, out);
}
